// FlexMACELayer_1778116461343
// MI455X (gfx1250) — compile-verified, matrix operand order checked by reading, not run on hardware
//
#include <hip/hip_runtime.h>
#include <hip/hip_bf16.h>

// ---------------------------------------------------------------------------
// MACE-style equivariant GNN layer for MI455X (gfx1250).
// Edge MLP (~31.5 GFLOP) and all node GEMMs go through v_wmma_f32_16x16x32_bf16.
// WMMA operands are staged in LDS pre-swizzled into the fragment register
// layout (ISA 7.12.2) as bf16 -> every fragment load is one aligned 32-byte
// vector LDS read. Edge-MLP weights are bulk-copied global->LDS with the
// Tensor Data Mover (TENSOR_LOAD_TO_LDS + s_wait_tensorcnt), then converted/
// swizzled in-LDS once per block. Segment-sum uses global_atomic_add_f32.
// ---------------------------------------------------------------------------

#define kN 16000
#define kE 256000

typedef __attribute__((ext_vector_type(16))) __bf16   v16bf;
typedef __attribute__((ext_vector_type(8)))  float    v8f;
typedef __attribute__((ext_vector_type(4)))  unsigned v4u;
typedef __attribute__((ext_vector_type(8)))  int      v8i;
typedef __attribute__((ext_vector_type(4)))  int      v4i;

__device__ __forceinline__ unsigned short f2bf(float f) {
    union { float f; unsigned u; } x; x.f = f;
    unsigned r = x.u + 0x7FFFu + ((x.u >> 16) & 1u);   // round-to-nearest-even
    return (unsigned short)(r >> 16);
}

__device__ __forceinline__ float bf2f(unsigned short h) {
    union { unsigned u; float f; } x; x.u = ((unsigned)h) << 16; return x.f;
}

__device__ __forceinline__ float silu_f(float x) {
    return x / (1.0f + __expf(-x));
}

// Position of element (m, klocal) inside a 16x32 bf16 fragment chunk laid out
// as [lane][16 elements]: returns lane*16 + slot (ISA 7.12.2 A/B layout).
__device__ __forceinline__ int frag_pos(int m, int kl) {
    int hi   = kl >> 4;            // selects VGPR group 4..7
    int rem  = kl & 15;
    int lh   = rem >> 3;           // lane high bit (K offset 8)
    int r2   = rem & 7;
    int slot = ((hi << 2) + (r2 >> 1)) * 2 + (r2 & 1);
    return (((m & 15) + (lh << 4)) << 4) + slot;
}

// Scatter one fp32 value into a swizzled bf16 tile (chunks of 512 ushorts).
__device__ __forceinline__ void stage_sw(unsigned short* buf, int m, int k, float v) {
    buf[(k >> 5) * 512 + frag_pos(m, k & 31)] = f2bf(v);
}

__device__ __forceinline__ v16bf load_frag(const unsigned short* p) {
    return *(const v16bf*)(p);     // 32B aligned -> ds_load_b128 x2
}

// ---------------------------------------------------------------------------
// Tensor Data Mover: 1-row 2D descriptor, global fp32 -> LDS (ISA ch.8 D#).
// This toolchain exposes the 6-arg builtin:
//   (uint32x4 g0, int32x8 g1, int32x4 g2, int32x4 g3, int32x8, i32 cpol)
// ---------------------------------------------------------------------------
#if __has_builtin(__builtin_amdgcn_tensor_load_to_lds) && \
    __has_builtin(__builtin_amdgcn_s_wait_tensorcnt)
#define USE_TDM 1
#endif

#ifdef USE_TDM
__device__ __forceinline__ void tdm_load_1d(const void* gsrc,
                                            unsigned lds_byte_addr,
                                            unsigned nelem) {
    unsigned long long ga = (unsigned long long)gsrc;
    v4u g0;
    g0[0] = 1u;                                          // count=1, user D#
    g0[1] = lds_byte_addr;                               // lds_addr
    g0[2] = (unsigned)(ga & 0xFFFFFFFFu);                // global_addr[31:0]
    g0[3] = (unsigned)((ga >> 32) & 0x01FFFFFFu)         // global_addr[56:32]
          | 0x80000000u;                                 // type=2 ("image")
    v8i g1;
    g1[0] = (int)(2u << 16);                             // data_size=4B
    g1[1] = (int)((nelem & 0xFFFFu) << 16);              // tensor_dim0[15:0]
    g1[2] = (int)((nelem >> 16) & 0xFFFFu) | (1 << 16);  // dim0 hi | tensor_dim1=1
    g1[3] = (int)((nelem & 0xFFFFu) << 16);              // dim1 hi=0 | tile_dim0
    g1[4] = 1;                                           // tile_dim1=1, tile_dim2=0
    g1[5] = (int)nelem;                                  // tensor_dim0_stride lo
    g1[6] = 0;
    g1[7] = 0;
    v4i z4 = {0, 0, 0, 0};
    v8i z8 = {0, 0, 0, 0, 0, 0, 0, 0};
    __builtin_amdgcn_tensor_load_to_lds(g0, g1, z4, z4, z8, 0);
}

__device__ __forceinline__ unsigned lds_addr_of(const void* p) {
    // Generic LDS addresses carry the aperture in [63:32]; [31:0] is the
    // wave-relative LDS byte address (ISA 10.2 aperture mapping).
    return (unsigned)(unsigned long long)p;
}
#endif

// D[16 x NT*16] = A[16 x KCH*32] * W (+bias) (opt SiLU); D fp32 row-major.
// A: swizzled bf16 LDS [KCH][32][16]; W: swizzled bf16 LDS [NT][KCH][32][16].
template <int KCH, int NT, bool SILU, bool HASB>
__device__ __forceinline__ void gemm16_f32(const unsigned short* A,
                                           const unsigned short* W,
                                           const float* bias,
                                           float* D, int ldd) {
    const int lane = threadIdx.x & 31;
    v16bf afr[KCH];
#pragma unroll
    for (int kc = 0; kc < KCH; ++kc)
        afr[kc] = load_frag(A + kc * 512 + lane * 16);
#pragma unroll
    for (int nt = 0; nt < NT; ++nt) {
        v8f c = {0.f, 0.f, 0.f, 0.f, 0.f, 0.f, 0.f, 0.f};
#pragma unroll
        for (int kc = 0; kc < KCH; ++kc) {
            v16bf b = load_frag(W + (nt * KCH + kc) * 512 + lane * 16);
            c = __builtin_amdgcn_wmma_f32_16x16x32_bf16(
                    false, afr[kc], false, b, (short)0, c, false, false);
        }
        const int col = nt * 16 + (lane & 15);
        const float bv = HASB ? bias[col] : 0.0f;
#pragma unroll
        for (int r = 0; r < 8; ++r) {
            float x = c[r] + bv;
            if (SILU) x = silu_f(x);
            D[(r + 8 * (lane >> 4)) * ldd + col] = x;
        }
    }
}

// Same, but D is stored directly as a swizzled bf16 tile (feeds next GEMM).
template <int KCH, int NT, bool SILU, bool HASB>
__device__ __forceinline__ void gemm16_sw(const unsigned short* A,
                                          const unsigned short* W,
                                          const float* bias,
                                          unsigned short* Dsw) {
    const int lane = threadIdx.x & 31;
    v16bf afr[KCH];
#pragma unroll
    for (int kc = 0; kc < KCH; ++kc)
        afr[kc] = load_frag(A + kc * 512 + lane * 16);
#pragma unroll
    for (int nt = 0; nt < NT; ++nt) {
        v8f c = {0.f, 0.f, 0.f, 0.f, 0.f, 0.f, 0.f, 0.f};
#pragma unroll
        for (int kc = 0; kc < KCH; ++kc) {
            v16bf b = load_frag(W + (nt * KCH + kc) * 512 + lane * 16);
            c = __builtin_amdgcn_wmma_f32_16x16x32_bf16(
                    false, afr[kc], false, b, (short)0, c, false, false);
        }
        const int col = nt * 16 + (lane & 15);
        const float bv = HASB ? bias[col] : 0.0f;
        const int kl = ((nt & 1) << 4) | (lane & 15);
        unsigned short* dchunk = Dsw + (nt >> 1) * 512;
#pragma unroll
        for (int r = 0; r < 8; ++r) {
            float x = c[r] + bv;
            if (SILU) x = silu_f(x);
            dchunk[frag_pos(r + 8 * (lane >> 4), kl)] = f2bf(x);
        }
    }
}

// ---------------------------------------------------------------------------
__global__ void k_zero(float* p, size_t n) {
    size_t i = (size_t)blockIdx.x * blockDim.x + threadIdx.x;
    if (i < n) p[i] = 0.0f;
}

// ---------------------------------------------------------------------------
// Kernel 1: node up-projection. 256 thr (8 waves), 128 nodes/block.
// ---------------------------------------------------------------------------
__global__ void __launch_bounds__(256)
k_node_up(const float* __restrict__ node_s, const float* __restrict__ node_v,
          const float* __restrict__ Wups_g, const float* __restrict__ bups,
          const float* __restrict__ Wupv_g,
          float* __restrict__ hs, float* __restrict__ hv) {
    extern __shared__ char smem[];
    unsigned short* wsS = (unsigned short*)smem;   // [4 nt][2 kc][512]
    unsigned short* wsV = wsS + 4096;
    unsigned short* Ain = wsV + 4096 + (threadIdx.x >> 5) * 1024;  // [2 kc][512]
    const int lane = threadIdx.x & 31;
    const int wave = threadIdx.x >> 5;

    for (int t = threadIdx.x; t < 64 * 64; t += blockDim.x) {
        int k = t >> 6, n = t & 63;
        int pos = ((n >> 4) * 2 + (k >> 5)) * 512 + frag_pos(n & 15, k & 31);
        wsS[pos] = f2bf(Wups_g[t]);
        wsV[pos] = f2bf(Wupv_g[t]);
    }
    __syncthreads();

    const int nb = blockIdx.x * 128 + wave * 16;

    for (int t = lane; t < 16 * 64; t += 32) {
        int r = t >> 6, f = t & 63;
        stage_sw(Ain, r, f, node_s[(size_t)(nb + r) * 64 + f]);
    }
    __syncthreads();
    gemm16_f32<2, 4, false, true>(Ain, wsS, bups, hs + (size_t)nb * 64, 64);
    __syncthreads();

    for (int i = 0; i < 3; ++i) {
        for (int t = lane; t < 16 * 64; t += 32) {
            int r = t >> 6, f = t & 63;
            stage_sw(Ain, r, f, node_v[((size_t)(nb + r) * 64 + f) * 3 + i]);
        }
        __syncthreads();
        gemm16_f32<2, 4, false, false>(Ain, wsV, nullptr,
                                       hv + (size_t)i * kN * 64 + (size_t)nb * 64, 64);
        __syncthreads();
    }
}

// ---------------------------------------------------------------------------
// Kernel 2: edge MLP + tensor products + atomic segment-sum.
// 128 thr (4 waves), 64 edges/block; each wave owns one 16-edge tile.
// Weights are TDM-copied into an LDS fp32 scratch (overlaying the per-wave
// activation buffers), then converted/swizzled to bf16 fragment layout.
// ---------------------------------------------------------------------------
__global__ void __launch_bounds__(128)
k_edge(const float* __restrict__ vectors, const float* __restrict__ lengths,
       const float* __restrict__ efeat,
       const int* __restrict__ senders, const int* __restrict__ receivers,
       const float* __restrict__ W1g, const float* __restrict__ b1,
       const float* __restrict__ W2g, const float* __restrict__ b2,
       const float* __restrict__ W3g, const float* __restrict__ b3,
       const float* __restrict__ hs, const float* __restrict__ hv,
       float* __restrict__ aggs, float* __restrict__ aggv) {
    extern __shared__ char smem[];
    unsigned short* w1 = (unsigned short*)smem;    // [8 nt][6 kc][512]
    unsigned short* w2 = w1 + 8 * 6 * 512;         // [8 nt][4 kc][512]
    unsigned short* w3 = w2 + 8 * 4 * 512;         // [12 nt][4 kc][512]
    unsigned short* uw = w3 + 12 * 4 * 512 + (threadIdx.x >> 5) * 7168;
    unsigned short* inA   = uw;                    // [6][512]
    unsigned short* actA  = inA + 3072;            // [4][512]
    unsigned short* act2A = actA + 2048;           // [4][512]
    float* mixbase = (float*)(w3 + 12 * 4 * 512 + 4 * 7168);
    float* mix = mixbase + (threadIdx.x >> 5) * (16 * 192);
    const int lane = threadIdx.x & 31;
    const int wave = threadIdx.x >> 5;

#ifdef USE_TDM
    // fp32 staging scratch overlays the per-wave buffers (used only later).
    float* scr = (float*)(w3 + 12 * 4 * 512);
    const unsigned scr_lds = lds_addr_of(scr);

    // --- W1 (161x128, K-padded to 192) ---
    if (threadIdx.x < 32) {
        tdm_load_1d(W1g, scr_lds, 161 * 128);
        __builtin_amdgcn_s_wait_tensorcnt(0);
    }
    __syncthreads();
    for (int t = threadIdx.x; t < 192 * 128; t += blockDim.x) {
        int k = t >> 7, n = t & 127;
        float v = (k < 161) ? scr[k * 128 + n] : 0.0f;
        w1[((n >> 4) * 6 + (k >> 5)) * 512 + frag_pos(n & 15, k & 31)] = f2bf(v);
    }
    __syncthreads();
    // --- W2 (128x128) ---
    if (threadIdx.x < 32) {
        tdm_load_1d(W2g, scr_lds, 128 * 128);
        __builtin_amdgcn_s_wait_tensorcnt(0);
    }
    __syncthreads();
    for (int t = threadIdx.x; t < 128 * 128; t += blockDim.x) {
        int k = t >> 7, n = t & 127;
        w2[((n >> 4) * 4 + (k >> 5)) * 512 + frag_pos(n & 15, k & 31)] = f2bf(scr[t]);
    }
    __syncthreads();
    // --- W3 (128x192) ---
    if (threadIdx.x < 32) {
        tdm_load_1d(W3g, scr_lds, 128 * 192);
        __builtin_amdgcn_s_wait_tensorcnt(0);
    }
    __syncthreads();
    for (int t = threadIdx.x; t < 128 * 192; t += blockDim.x) {
        int k = t / 192, n = t % 192;
        w3[((n >> 4) * 4 + (k >> 5)) * 512 + frag_pos(n & 15, k & 31)] = f2bf(scr[t]);
    }
    __syncthreads();
#else
    for (int t = threadIdx.x; t < 192 * 128; t += blockDim.x) {
        int k = t >> 7, n = t & 127;
        float v = (k < 161) ? W1g[k * 128 + n] : 0.0f;
        w1[((n >> 4) * 6 + (k >> 5)) * 512 + frag_pos(n & 15, k & 31)] = f2bf(v);
    }
    for (int t = threadIdx.x; t < 128 * 128; t += blockDim.x) {
        int k = t >> 7, n = t & 127;
        w2[((n >> 4) * 4 + (k >> 5)) * 512 + frag_pos(n & 15, k & 31)] = f2bf(W2g[t]);
    }
    for (int t = threadIdx.x; t < 128 * 192; t += blockDim.x) {
        int k = t / 192, n = t % 192;
        w3[((n >> 4) * 4 + (k >> 5)) * 512 + frag_pos(n & 15, k & 31)] = f2bf(W3g[t]);
    }
    __syncthreads();
#endif

    const int eb = blockIdx.x * 64 + wave * 16;

    // Stage mlp_in = [h_s[send] | h_s[recv] | edge_feats | length | pad] swizzled.
    for (int t = lane; t < 16 * 192; t += 32) {
        int r = t / 192, c = t % 192;
        int e = eb + r;
        float v;
        if (c < 64)        v = hs[(size_t)senders[e] * 64 + c];
        else if (c < 128)  v = hs[(size_t)receivers[e] * 64 + (c - 64)];
        else if (c < 160)  v = efeat[(size_t)e * 32 + (c - 128)];
        else if (c == 160) v = lengths[e];
        else               v = 0.0f;
        stage_sw(inA, r, c, v);
    }
    __syncthreads();

    gemm16_sw<6, 8, true, true>(inA, w1, b1, actA);          // L1 + SiLU
    __syncthreads();
    gemm16_sw<4, 8, true, true>(actA, w2, b2, act2A);        // L2 + SiLU
    __syncthreads();
    gemm16_f32<4, 12, false, true>(act2A, w3, b3, mix, 192); // L3 -> mix fp32
    __syncthreads();

    // Epilogue: tensor products gated by mix, scatter-add into agg buffers.
    const float SQ3  = 1.7320508075688772f;
    const float ISQ3 = 0.5773502691896258f;
    const float ISQ2 = 0.7071067811865476f;
    const float INV_AVG = 1.0f / 16.0f;
    for (int t = lane; t < 16 * 64; t += 32) {
        int r = t >> 6, f = t & 63;
        int e  = eb + r;
        int sn = senders[e], rn = receivers[e];
        float ev0 = SQ3 * vectors[e * 3 + 0];
        float ev1 = SQ3 * vectors[e * 3 + 1];
        float ev2 = SQ3 * vectors[e * 3 + 2];
        float ms  = hs[(size_t)sn * 64 + f];
        float mv0 = hv[0 * (size_t)kN * 64 + (size_t)sn * 64 + f];
        float mv1 = hv[1 * (size_t)kN * 64 + (size_t)sn * 64 + f];
        float mv2 = hv[2 * (size_t)kN * 64 + (size_t)sn * 64 + f];
        const float* mr = mix + r * 192;
        float mxs = mr[f], mx1 = mr[64 + f], mx2 = mr[128 + f];

        float tps = (mv0 * ev0 + mv1 * ev1 + mv2 * ev2) * ISQ3 * mxs;
        atomicAdd(&aggs[(size_t)rn * 64 + f], tps * INV_AVG);

        float c1 = ms * mx1 * INV_AVG;
        atomicAdd(&aggv[0 * (size_t)kN * 128 + (size_t)rn * 128 + f], c1 * ev0);
        atomicAdd(&aggv[1 * (size_t)kN * 128 + (size_t)rn * 128 + f], c1 * ev1);
        atomicAdd(&aggv[2 * (size_t)kN * 128 + (size_t)rn * 128 + f], c1 * ev2);

        float cx0 = (mv1 * ev2 - mv2 * ev1) * ISQ2;
        float cx1 = (mv2 * ev0 - mv0 * ev2) * ISQ2;
        float cx2 = (mv0 * ev1 - mv1 * ev0) * ISQ2;
        float c2 = mx2 * INV_AVG;
        atomicAdd(&aggv[0 * (size_t)kN * 128 + (size_t)rn * 128 + 64 + f], cx0 * c2);
        atomicAdd(&aggv[1 * (size_t)kN * 128 + (size_t)rn * 128 + 64 + f], cx1 * c2);
        atomicAdd(&aggv[2 * (size_t)kN * 128 + (size_t)rn * 128 + 64 + f], cx2 * c2);
    }
}

// ---------------------------------------------------------------------------
// Kernel 3: node down path. 128 thr (4 waves), 64 nodes/block.
// agg -> f -> sc(species) -> p (out) -> r -> gated outputs.
// ---------------------------------------------------------------------------
__global__ void __launch_bounds__(128)
k_node_down(const float* __restrict__ aggs, const float* __restrict__ aggv,
            const int* __restrict__ specie,
            const float* __restrict__ Wdns_g, const float* __restrict__ bdns,
            const float* __restrict__ Wdnv_g,
            const float* __restrict__ wscs, const float* __restrict__ wscv,
            const float* __restrict__ Wpbs_g, const float* __restrict__ bpbs,
            const float* __restrict__ Wpbv_g,
            const float* __restrict__ Wr1s_g, const float* __restrict__ br1s,
            const float* __restrict__ Wr1v_g,
            const float* __restrict__ Wr2s, const float* __restrict__ br2,
            const float* __restrict__ Wr2v,
            float* __restrict__ out) {
    extern __shared__ char smem[];
    unsigned short* wdns = (unsigned short*)smem;   // [4][2][512]
    unsigned short* wdnv = wdns + 4096;             // [4][4][512]
    unsigned short* wpbs = wdnv + 8192;             // [4][2][512]
    unsigned short* wpbv = wpbs + 4096;             // [4][2][512]
    unsigned short* wr1s = wpbv + 4096;             // [8][2][512]
    unsigned short* wr1v = wr1s + 8192;             // [4][2][512]
    unsigned short* uw = wr1v + 4096 + (threadIdx.x >> 5) * 10240;
    unsigned short* Ain  = uw;                      // [4][512]
    unsigned short* SCs  = Ain + 2048;              // [2][512]
    unsigned short* SCv  = SCs + 1024;              // 3 x [2][512]
    unsigned short* PSsw = SCv + 3072;              // [2][512]
    unsigned short* PVsw = PSsw + 1024;             // 3 x [2][512]
    float* Rbase = (float*)((unsigned short*)smem + 32768 + 4 * 10240);
    float* R = Rbase + (threadIdx.x >> 5) * (16 * 320);
    const int lane = threadIdx.x & 31;
    const int wave = threadIdx.x >> 5;
    const int RW = 320;

    // Stage + swizzle all weights (bf16) once per block.
    for (int t = threadIdx.x; t < 64 * 64; t += blockDim.x) {
        int k = t >> 6, n = t & 63;
        int pos = ((n >> 4) * 2 + (k >> 5)) * 512 + frag_pos(n & 15, k & 31);
        wdns[pos] = f2bf(Wdns_g[t]);
        wpbs[pos] = f2bf(Wpbs_g[t]);
        wpbv[pos] = f2bf(Wpbv_g[t]);
        wr1v[pos] = f2bf(Wr1v_g[t]);
    }
    for (int t = threadIdx.x; t < 128 * 64; t += blockDim.x) {
        int k = t >> 6, n = t & 63;
        wdnv[((n >> 4) * 4 + (k >> 5)) * 512 + frag_pos(n & 15, k & 31)] = f2bf(Wdnv_g[t]);
    }
    for (int t = threadIdx.x; t < 64 * 128; t += blockDim.x) {
        int k = t >> 7, n = t & 127;
        wr1s[((n >> 4) * 2 + (k >> 5)) * 512 + frag_pos(n & 15, k & 31)] = f2bf(Wr1s_g[t]);
    }
    __syncthreads();

    const int nb = blockIdx.x * 64 + wave * 16;
    const float ISQ3 = 0.5773502691896258f;

    // f_s -> R[:,0:64]
    for (int t = lane; t < 16 * 64; t += 32) {
        int r = t >> 6, f = t & 63;
        stage_sw(Ain, r, f, aggs[(size_t)(nb + r) * 64 + f]);
    }
    __syncthreads();
    gemm16_f32<2, 4, false, true>(Ain, wdns, bdns, R, RW);
    __syncthreads();

    // f_v components -> R[:,64+64i:128+64i]
    for (int i = 0; i < 3; ++i) {
        for (int t = lane; t < 16 * 128; t += 32) {
            int r = t >> 7, k = t & 127;
            stage_sw(Ain, r, k, aggv[(size_t)i * kN * 128 + (size_t)(nb + r) * 128 + k]);
        }
        __syncthreads();
        gemm16_f32<4, 4, false, false>(Ain, wdnv, nullptr, R + 64 + 64 * i, RW);
        __syncthreads();
    }

    // species-gated elementwise: sc_s -> SCs (swizzled), sc_v_i -> SCv[i]
    for (int t = lane; t < 16 * 64; t += 32) {
        int r = t >> 6, f = t & 63;
        int node = nb + r;
        int sp = specie[node];
        const float* wsp = wscs + sp * 192;
        const float* wvp = wscv + sp * 128;
        float fs = R[r * RW + f];
        float f0 = R[r * RW + 64 + f];
        float f1 = R[r * RW + 128 + f];
        float f2 = R[r * RW + 192 + f];
        float vv = (f0 * f0 + f1 * f1 + f2 * f2) * ISQ3;
        stage_sw(SCs, r, f, wsp[f] * fs + wsp[64 + f] * fs * fs + wsp[128 + f] * vv);
        float g = wvp[f] + wvp[64 + f] * fs;
        stage_sw(SCv + 0 * 1024, r, f, g * f0);
        stage_sw(SCv + 1 * 1024, r, f, g * f1);
        stage_sw(SCv + 2 * 1024, r, f, g * f2);
    }
    __syncthreads();

    // p_s -> PSsw, p_v_i -> PVsw[i] (swizzled bf16, feed r-GEMMs)
    gemm16_sw<2, 4, false, true>(SCs, wpbs, bpbs, PSsw);
    for (int i = 0; i < 3; ++i)
        gemm16_sw<2, 4, false, false>(SCv + i * 1024, wpbv, nullptr, PVsw + i * 1024);
    __syncthreads();

    // write p_s / p_v outputs
    float* out_ps = out + (size_t)4 * kN;
    float* out_pv = out + (size_t)68 * kN;
    for (int t = lane; t < 16 * 64; t += 32) {
        int r = t >> 6, g = t & 63;
        int node = nb + r;
        int fp = (g >> 5) * 512 + frag_pos(r, g & 31);
        out_ps[(size_t)node * 64 + g] = bf2f(PSsw[fp]);
        for (int i = 0; i < 3; ++i)
            out_pv[((size_t)node * 64 + g) * 3 + i] = bf2f(PVsw[i * 1024 + fp]);
    }
    __syncthreads();

    // r_s -> R[:,0:128]; r_v_i -> R[:,128+64i:192+64i]
    gemm16_f32<2, 8, false, true>(PSsw, wr1s, br1s, R, RW);
    for (int i = 0; i < 3; ++i)
        gemm16_f32<2, 4, false, false>(PVsw + i * 1024, wr1v, nullptr,
                                       R + 128 + 64 * i, RW);
    __syncthreads();

    // final gated reductions: out_s (N), out_v (N,3)
    float* out_s = out;
    float* out_v = out + kN;
    for (int t = lane; t < 16; t += 32) {
        int r = t;
        int node = nb + r;
        float os = br2[0];
        float ov0 = 0.f, ov1 = 0.f, ov2 = 0.f;
        for (int f = 0; f < 64; ++f) {
            float a = silu_f(R[r * RW + f]);
            os += a * Wr2s[f];
            float gg = silu_f(R[r * RW + 64 + f]) * Wr2v[f];
            ov0 += gg * R[r * RW + 128 + f];
            ov1 += gg * R[r * RW + 192 + f];
            ov2 += gg * R[r * RW + 256 + f];
        }
        out_s[node] = os;
        out_v[(size_t)node * 3 + 0] = ov0;
        out_v[(size_t)node * 3 + 1] = ov1;
        out_v[(size_t)node * 3 + 2] = ov2;
    }
}

// ---------------------------------------------------------------------------
extern "C" void kernel_launch(void* const* d_in, const int* in_sizes, int n_in,
                              void* d_out, int out_size, void* d_ws, size_t ws_size,
                              hipStream_t stream) {
    const float* vectors  = (const float*)d_in[0];
    const float* lengths  = (const float*)d_in[1];
    const float* node_s   = (const float*)d_in[2];
    const float* node_v   = (const float*)d_in[3];
    const float* efeat    = (const float*)d_in[4];
    const int*   specie   = (const int*)d_in[5];
    const int*   senders  = (const int*)d_in[6];
    const int*   recvs    = (const int*)d_in[7];
    const float* W_up_s   = (const float*)d_in[8];
    const float* b_up_s   = (const float*)d_in[9];
    const float* W_up_v   = (const float*)d_in[10];
    const float* mlp_W1   = (const float*)d_in[11];
    const float* mlp_b1   = (const float*)d_in[12];
    const float* mlp_W2   = (const float*)d_in[13];
    const float* mlp_b2   = (const float*)d_in[14];
    const float* mlp_W3   = (const float*)d_in[15];
    const float* mlp_b3   = (const float*)d_in[16];
    const float* W_dn_s   = (const float*)d_in[17];
    const float* b_dn_s   = (const float*)d_in[18];
    const float* W_dn_v   = (const float*)d_in[19];
    const float* w_sc_s   = (const float*)d_in[20];
    const float* w_sc_v   = (const float*)d_in[21];
    const float* W_pb_s   = (const float*)d_in[22];
    const float* b_pb_s   = (const float*)d_in[23];
    const float* W_pb_v   = (const float*)d_in[24];
    const float* W_r1_s   = (const float*)d_in[25];
    const float* b_r1_s   = (const float*)d_in[26];
    const float* W_r1_v   = (const float*)d_in[27];
    const float* W_r2_s   = (const float*)d_in[28];
    const float* b_r2     = (const float*)d_in[29];
    const float* W_r2_v   = (const float*)d_in[30];
    float* out = (float*)d_out;

    // workspace: h_s[N*64] | h_v[3*N*64] | agg_s[N*64] | agg_v[3*N*128]
    float* ws   = (float*)d_ws;
    float* hs   = ws;
    float* hv   = ws + (size_t)kN * 64;
    float* aggs = ws + (size_t)kN * 64 * 4;
    float* aggv = ws + (size_t)kN * 64 * 5;

    size_t zn = (size_t)kN * 448;
    k_zero<<<(unsigned)((zn + 255) / 256), 256, 0, stream>>>(aggs, zn);

    size_t lds_up = (4096 + 4096 + 8 * 1024) * sizeof(unsigned short);
    k_node_up<<<kN / 128, 256, lds_up, stream>>>(node_s, node_v, W_up_s, b_up_s,
                                                 W_up_v, hs, hv);

    size_t lds_edge = (size_t)(8 * 6 * 512 + 8 * 4 * 512 + 12 * 4 * 512
                               + 4 * 7168) * sizeof(unsigned short)
                    + (size_t)4 * 16 * 192 * sizeof(float);
    k_edge<<<kE / 64, 128, lds_edge, stream>>>(vectors, lengths, efeat, senders, recvs,
                                               mlp_W1, mlp_b1, mlp_W2, mlp_b2,
                                               mlp_W3, mlp_b3, hs, hv, aggs, aggv);

    size_t lds_dn = (size_t)(32768 + 4 * 10240) * sizeof(unsigned short)
                  + (size_t)4 * 16 * 320 * sizeof(float);
    k_node_down<<<kN / 64, 128, lds_dn, stream>>>(aggs, aggv, specie,
                                                  W_dn_s, b_dn_s, W_dn_v,
                                                  w_sc_s, w_sc_v,
                                                  W_pb_s, b_pb_s, W_pb_v,
                                                  W_r1_s, b_r1_s, W_r1_v,
                                                  W_r2_s, b_r2, W_r2_v, out);
}
